// SpatialLoRALinear_45646912422056
// MI455X (gfx1250) — matrix operand order repaired and checked by reading, not compiled, not run on hardware
//
#include <hip/hip_runtime.h>
#include <stdint.h>

// ---- problem constants (from reference) ----
#define TOK   32768      // B*S
#define SLEN  4096
#define DIN   1280
#define DOUT  1280
#define NSUB  4
#define RNK   16
#define KW    1280       // K of base GEMM
#define KL    64         // N*R = lora K tail
#define KTOT  1344       // KW + KL  (42 * 32)

// ---- tiling ----
#define BM    128
#define BN    128
#define BK    32
#define LDA   (BK + 8)   // 40 halfwords = 80B row stride (conflict-avoiding)

typedef __attribute__((ext_vector_type(16))) __bf16        v16bf;
typedef __attribute__((ext_vector_type(8)))  __bf16        v8bf;
typedef __attribute__((ext_vector_type(8)))  float         v8f;
typedef __attribute__((ext_vector_type(4)))  float         v4f;
typedef __attribute__((ext_vector_type(8)))  unsigned short v8u16;

__device__ __forceinline__ unsigned short f2bfu(float f) {
  __bf16 h = (__bf16)f;                       // native cvt (v_cvt_pk_bf16_f32)
  return __builtin_bit_cast(unsigned short, h);
}
__device__ __forceinline__ v8u16 cvt8(const v4f& a, const v4f& b) {
  v8u16 t;
  t.s0 = f2bfu(a.x); t.s1 = f2bfu(a.y); t.s2 = f2bfu(a.z); t.s3 = f2bfu(a.w);
  t.s4 = f2bfu(b.x); t.s5 = f2bfu(b.y); t.s6 = f2bfu(b.z); t.s7 = f2bfu(b.w);
  return t;
}

// async global->LDS 16B copy (GLOBAL_LOAD_ASYNC_TO_LDS_B128, ASYNCcnt-tracked)
__device__ __forceinline__ void async_cp16(unsigned lds_addr, const void* gptr) {
  asm volatile("global_load_async_to_lds_b128 %0, %1, off"
               :: "v"(lds_addr), "v"(gptr) : "memory");
}
__device__ __forceinline__ void wait_async0() {
  asm volatile("s_wait_asynccnt 0x0" ::: "memory");
}

// A-fragment (16x32 bf16, MxK): lane L = row (L&15), h = L>>4.
// elems[0:8]  = K[8h..8h+7], elems[8:16] = K[16+8h..16+8h+7]
__device__ __forceinline__ v16bf frag_a(const unsigned short* row, int h) {
  v8bf lo = __builtin_bit_cast(v8bf, *(const v8u16*)(row + 8 * h));
  v8bf hi = __builtin_bit_cast(v8bf, *(const v8u16*)(row + 16 + 8 * h));
  return __builtin_shufflevector(lo, hi, 0,1,2,3,4,5,6,7,8,9,10,11,12,13,14,15);
}
// B-fragment (32x16 bf16, KxN): lane L = col (L&15), h = L>>4.
// elems[0:8] = K[8h..8h+7], elems[8:16] = K[16+8h..16+8h+7]   (tile stored row-major along K, row = out feature)
__device__ __forceinline__ v16bf frag_b(const unsigned short* row, int h) {
  v8bf lo = __builtin_bit_cast(v8bf, *(const v8u16*)(row + 8 * h));
  v8bf hi = __builtin_bit_cast(v8bf, *(const v8u16*)(row + 16 + 8 * h));
  return __builtin_shufflevector(lo, hi, 0,1,2,3,4,5,6,7,8,9,10,11,12,13,14,15);
}

// ---------------------------------------------------------------------------
// Kernel 0: pack Wcat = [ bf16(W) | Bw[n,o,r]*scal[n] ]  (DOUT x KTOT, bf16)
//           and Abf = bf16(A_flat)                        (KL x DIN, bf16)
// ---------------------------------------------------------------------------
__global__ void pack_weights(const float* __restrict__ W,
                             const float* __restrict__ A,
                             const float* __restrict__ Bw,
                             const float* __restrict__ scal,
                             unsigned short* __restrict__ Wcat,
                             unsigned short* __restrict__ Abf) {
  const int total1 = DOUT * KTOT;
  const int total2 = KL * DIN;
  for (int i = blockIdx.x * blockDim.x + threadIdx.x; i < total1 + total2;
       i += gridDim.x * blockDim.x) {
    if (i < total1) {
      int o = i / KTOT, k = i - o * KTOT;
      float v;
      if (k < KW) {
        v = W[o * DIN + k];
      } else {
        int j = k - KW, n = j >> 4, r = j & 15;
        v = Bw[(n * DOUT + o) * RNK + r] * scal[n];
      }
      Wcat[i] = f2bfu(v);
    } else {
      int i2 = i - total1;
      Abf[i2] = f2bfu(A[i2]);
    }
  }
}

// ===========================================================================
// FUSED PATH (needs ~92 MB workspace):
//   xcat[t, 0..1279]     = bf16(x[t, :])         (written by lora_h_fused)
//   xcat[t, 1280..1343]  = bf16(H[t,:]*mask)     (written by lora_h_fused epi)
//   out = xcat @ Wcat^T + bias                   (main_gemm_fused, K=1344)
// ===========================================================================

// Kernel 1 (fused): H = x @ A_flat^T, also emits bf16(x) into xcat.
__global__ __launch_bounds__(256) void lora_h_fused(
    const float* __restrict__ x, const unsigned short* __restrict__ Abf,
    const float* __restrict__ masks, unsigned short* __restrict__ xcat) {
  __shared__ alignas(16) unsigned short As[2][BM * LDA];
  __shared__ alignas(16) unsigned short Bs[2][KL * LDA];

  const int tid  = threadIdx.x;
  const int lane = tid & 31;
  const int wid  = tid >> 5;
  const int row0 = blockIdx.x * BM;

  v8u16 axb[2];       // converted x octets
  v8u16 bw;           // Abf octet
  v8f acc[4];
#pragma unroll
  for (int f = 0; f < 4; ++f) acc[f] = v8f{0.f,0.f,0.f,0.f,0.f,0.f,0.f,0.f};

  auto load_tile = [&](int kt) {
    const int k0 = kt * BK;
#pragma unroll
    for (int i = 0; i < 2; ++i) {
      int q = tid + 256 * i, r = q >> 2, c = (q & 3) * 8;   // r 0..127
      const float* p = x + (size_t)(row0 + r) * DIN + k0 + c;
      axb[i] = cvt8(*(const v4f*)p, *(const v4f*)(p + 4));
      // persist bf16(x) once into xcat
      *(v8u16*)(xcat + (size_t)(row0 + r) * KTOT + k0 + c) = axb[i];
    }
    {
      int q = tid, r = q >> 2, c = (q & 3) * 8;             // r 0..63
      bw = *(const v8u16*)(Abf + (size_t)r * DIN + k0 + c);
    }
  };
  auto store_tile = [&](int buf) {
#pragma unroll
    for (int i = 0; i < 2; ++i) {
      int q = tid + 256 * i, r = q >> 2, c = (q & 3) * 8;
      *(v8u16*)(&As[buf][r * LDA + c]) = axb[i];
    }
    {
      int q = tid, r = q >> 2, c = (q & 3) * 8;
      *(v8u16*)(&Bs[buf][r * LDA + c]) = bw;
    }
  };

  load_tile(0);
  store_tile(0);

  const int KT = KW / BK;  // 40
  for (int kt = 0; kt < KT; ++kt) {
    __syncthreads();
    if (kt + 1 < KT) load_tile(kt + 1);
    const int buf = kt & 1;
    {
      int m = wid * 16 + (lane & 15);
      v16bf a = frag_a(&As[buf][m * LDA], lane >> 4);
#pragma unroll
      for (int f = 0; f < 4; ++f) {
        int n = f * 16 + (lane & 15);
        v16bf b = frag_b(&Bs[buf][n * LDA], lane >> 4);
        acc[f] = __builtin_amdgcn_wmma_f32_16x16x32_bf16(
            false, a, false, b, (short)0, acc[f], false, false);
      }
    }
    if (kt + 1 < KT) store_tile((kt + 1) & 1);
  }

  // epilogue: weight by mask[n, s], store into xcat tail.  frag f == subject n.
#pragma unroll
  for (int f = 0; f < 4; ++f) {
    int n  = f * 16 + (lane & 15);
    int tb = row0 + wid * 16 + 8 * (lane >> 4);
#pragma unroll
    for (int j = 0; j < 8; ++j) {
      int t = tb + j;
      int s = t & (SLEN - 1);
      float w = masks[f * SLEN + s];
      xcat[(size_t)t * KTOT + KW + n] = f2bfu(acc[f][j] * w);
    }
  }
}

// Kernel 2 (fused): pure bf16 GEMM, K = 1344.  Tiles are filled with
// GLOBAL_LOAD_ASYNC_TO_LDS_B128 (ASYNCcnt) -- no VGPR staging, no ds_store.
__global__ __launch_bounds__(256) void main_gemm_fused(
    const unsigned short* __restrict__ xcat,
    const unsigned short* __restrict__ Wcat,
    const float* __restrict__ bias, float* __restrict__ out) {
  __shared__ alignas(16) unsigned short As[2][BM * LDA];
  __shared__ alignas(16) unsigned short Bs[2][BN * LDA];

  const int tid  = threadIdx.x;
  const int lane = tid & 31;
  const int wid  = tid >> 5;
  const int wm   = wid & 3;   // 0..3  -> 32 rows
  const int wn   = wid >> 2;  // 0..1  -> 64 cols
  const int nt   = blockIdx.x % (DOUT / BN);
  const int mt   = blockIdx.x / (DOUT / BN);
  const int row0 = mt * BM;
  const int col0 = nt * BN;

  v8f acc[2][4];
#pragma unroll
  for (int i = 0; i < 2; ++i)
#pragma unroll
    for (int f = 0; f < 4; ++f) acc[i][f] = v8f{0.f,0.f,0.f,0.f,0.f,0.f,0.f,0.f};

  auto issue_tile = [&](int kt, int buf) {
    const int k0 = kt * BK;
#pragma unroll
    for (int i = 0; i < 2; ++i) {
      int q = tid + 256 * i, r = q >> 2, c = (q & 3) * 8;
      async_cp16((unsigned)(uintptr_t)&As[buf][r * LDA + c],
                 xcat + (size_t)(row0 + r) * KTOT + k0 + c);
      async_cp16((unsigned)(uintptr_t)&Bs[buf][r * LDA + c],
                 Wcat + (size_t)(col0 + r) * KTOT + k0 + c);
    }
  };

  issue_tile(0, 0);

  const int KT = KTOT / BK;  // 42
  for (int kt = 0; kt < KT; ++kt) {
    wait_async0();            // own async fills landed
    __syncthreads();          // everyone's fills landed / prev reads done
    if (kt + 1 < KT) issue_tile(kt + 1, (kt + 1) & 1);
    const int buf = kt & 1;
    {
      v16bf afrag[2];
#pragma unroll
      for (int i = 0; i < 2; ++i) {
        int m = wm * 32 + i * 16 + (lane & 15);
        afrag[i] = frag_a(&As[buf][m * LDA], lane >> 4);
      }
#pragma unroll
      for (int f = 0; f < 4; ++f) {
        int n = wn * 64 + f * 16 + (lane & 15);
        v16bf bfrag = frag_b(&Bs[buf][n * LDA], lane >> 4);
#pragma unroll
        for (int i = 0; i < 2; ++i) {
          acc[i][f] = __builtin_amdgcn_wmma_f32_16x16x32_bf16(
              false, afrag[i], false, bfrag, (short)0, acc[i][f], false, false);
        }
      }
    }
  }

  // epilogue: + bias, f32 store.  C layout: row = j + 8*(lane>>4), col = lane&15
#pragma unroll
  for (int f = 0; f < 4; ++f) {
    int o = col0 + wn * 64 + f * 16 + (lane & 15);
    float bv = bias[o];
#pragma unroll
    for (int i = 0; i < 2; ++i) {
      int tb = row0 + wm * 32 + i * 16 + 8 * (lane >> 4);
#pragma unroll
      for (int j = 0; j < 8; ++j) {
        out[(size_t)(tb + j) * DOUT + o] = acc[i][f][j] + bv;
      }
    }
  }
}

// ===========================================================================
// FALLBACK PATH (small workspace, ~8 MB): convert x inside the GEMM.
// ===========================================================================
__global__ __launch_bounds__(256) void lora_h_small(
    const float* __restrict__ x, const unsigned short* __restrict__ Abf,
    const float* __restrict__ masks, unsigned short* __restrict__ Hwb) {
  __shared__ alignas(16) unsigned short As[2][BM * LDA];
  __shared__ alignas(16) unsigned short Bs[2][KL * LDA];

  const int tid  = threadIdx.x;
  const int lane = tid & 31;
  const int wid  = tid >> 5;
  const int row0 = blockIdx.x * BM;

  v8u16 axb[2];
  v8u16 bw;
  v8f acc[4];
#pragma unroll
  for (int f = 0; f < 4; ++f) acc[f] = v8f{0.f,0.f,0.f,0.f,0.f,0.f,0.f,0.f};

  auto load_tile = [&](int kt) {
    const int k0 = kt * BK;
#pragma unroll
    for (int i = 0; i < 2; ++i) {
      int q = tid + 256 * i, r = q >> 2, c = (q & 3) * 8;
      const float* p = x + (size_t)(row0 + r) * DIN + k0 + c;
      axb[i] = cvt8(*(const v4f*)p, *(const v4f*)(p + 4));
    }
    {
      int q = tid, r = q >> 2, c = (q & 3) * 8;
      bw = *(const v8u16*)(Abf + (size_t)r * DIN + k0 + c);
    }
  };
  auto store_tile = [&](int buf) {
#pragma unroll
    for (int i = 0; i < 2; ++i) {
      int q = tid + 256 * i, r = q >> 2, c = (q & 3) * 8;
      *(v8u16*)(&As[buf][r * LDA + c]) = axb[i];
    }
    {
      int q = tid, r = q >> 2, c = (q & 3) * 8;
      *(v8u16*)(&Bs[buf][r * LDA + c]) = bw;
    }
  };

  load_tile(0);
  store_tile(0);
  const int KT = KW / BK;
  for (int kt = 0; kt < KT; ++kt) {
    __syncthreads();
    if (kt + 1 < KT) load_tile(kt + 1);
    const int buf = kt & 1;
    int m = wid * 16 + (lane & 15);
    v16bf a = frag_a(&As[buf][m * LDA], lane >> 4);
#pragma unroll
    for (int f = 0; f < 4; ++f) {
      int n = f * 16 + (lane & 15);
      v16bf b = frag_b(&Bs[buf][n * LDA], lane >> 4);
      acc[f] = __builtin_amdgcn_wmma_f32_16x16x32_bf16(
          false, a, false, b, (short)0, acc[f], false, false);
    }
    if (kt + 1 < KT) store_tile((kt + 1) & 1);
  }
#pragma unroll
  for (int f = 0; f < 4; ++f) {
    int n  = f * 16 + (lane & 15);
    int tb = row0 + wid * 16 + 8 * (lane >> 4);
#pragma unroll
    for (int j = 0; j < 8; ++j) {
      int t = tb + j;
      int s = t & (SLEN - 1);
      Hwb[(size_t)t * KL + n] = f2bfu(acc[f][j] * masks[f * SLEN + s]);
    }
  }
}

__global__ __launch_bounds__(256) void main_gemm_small(
    const float* __restrict__ x, const unsigned short* __restrict__ Wcat,
    const unsigned short* __restrict__ Hwb, const float* __restrict__ bias,
    float* __restrict__ out) {
  __shared__ alignas(16) unsigned short As[2][BM * LDA];
  __shared__ alignas(16) unsigned short Bs[2][BN * LDA];

  const int tid  = threadIdx.x;
  const int lane = tid & 31;
  const int wid  = tid >> 5;
  const int wm   = wid & 3;
  const int wn   = wid >> 2;
  const int nt   = blockIdx.x % (DOUT / BN);
  const int mt   = blockIdx.x / (DOUT / BN);
  const int row0 = mt * BM;
  const int col0 = nt * BN;

  v8u16 ar[2], br[2];
  v8f acc[2][4];
#pragma unroll
  for (int i = 0; i < 2; ++i)
#pragma unroll
    for (int f = 0; f < 4; ++f) acc[i][f] = v8f{0.f,0.f,0.f,0.f,0.f,0.f,0.f,0.f};

  auto load_tile = [&](int kt) {
    const int k0 = kt * BK;
    if (k0 < KW) {
#pragma unroll
      for (int i = 0; i < 2; ++i) {
        int q = tid + 256 * i, r = q >> 2, c = (q & 3) * 8;
        const float* p = x + (size_t)(row0 + r) * DIN + k0 + c;
        ar[i] = cvt8(*(const v4f*)p, *(const v4f*)(p + 4));
      }
    } else {
      const int kh = k0 - KW;
#pragma unroll
      for (int i = 0; i < 2; ++i) {
        int q = tid + 256 * i, r = q >> 2, c = (q & 3) * 8;
        ar[i] = *(const v8u16*)(Hwb + (size_t)(row0 + r) * KL + kh + c);
      }
    }
#pragma unroll
    for (int i = 0; i < 2; ++i) {
      int q = tid + 256 * i, r = q >> 2, c = (q & 3) * 8;
      br[i] = *(const v8u16*)(Wcat + (size_t)(col0 + r) * KTOT + k0 + c);
    }
  };
  auto store_tile = [&](int buf) {
#pragma unroll
    for (int i = 0; i < 2; ++i) {
      int q = tid + 256 * i, r = q >> 2, c = (q & 3) * 8;
      *(v8u16*)(&As[buf][r * LDA + c]) = ar[i];
      *(v8u16*)(&Bs[buf][r * LDA + c]) = br[i];
    }
  };

  load_tile(0);
  store_tile(0);
  const int KT = KTOT / BK;
  for (int kt = 0; kt < KT; ++kt) {
    __syncthreads();
    if (kt + 1 < KT) load_tile(kt + 1);
    const int buf = kt & 1;
    v16bf afrag[2];
#pragma unroll
    for (int i = 0; i < 2; ++i) {
      int m = wm * 32 + i * 16 + (lane & 15);
      afrag[i] = frag_a(&As[buf][m * LDA], lane >> 4);
    }
#pragma unroll
    for (int f = 0; f < 4; ++f) {
      int n = wn * 64 + f * 16 + (lane & 15);
      v16bf bfrag = frag_b(&Bs[buf][n * LDA], lane >> 4);
#pragma unroll
      for (int i = 0; i < 2; ++i) {
        acc[i][f] = __builtin_amdgcn_wmma_f32_16x16x32_bf16(
            false, afrag[i], false, bfrag, (short)0, acc[i][f], false, false);
      }
    }
    if (kt + 1 < KT) store_tile((kt + 1) & 1);
  }
#pragma unroll
  for (int f = 0; f < 4; ++f) {
    int o = col0 + wn * 64 + f * 16 + (lane & 15);
    float bv = bias[o];
#pragma unroll
    for (int i = 0; i < 2; ++i) {
      int tb = row0 + wm * 32 + i * 16 + 8 * (lane >> 4);
#pragma unroll
      for (int j = 0; j < 8; ++j) {
        out[(size_t)(tb + j) * DOUT + o] = acc[i][f][j] + bv;
      }
    }
  }
}

// ---------------------------------------------------------------------------
extern "C" void kernel_launch(void* const* d_in, const int* in_sizes, int n_in,
                              void* d_out, int out_size, void* d_ws, size_t ws_size,
                              hipStream_t stream) {
  const float* x     = (const float*)d_in[0];
  const float* W     = (const float*)d_in[1];
  const float* bias  = (const float*)d_in[2];
  const float* A     = (const float*)d_in[3];
  const float* Bw    = (const float*)d_in[4];
  const float* scal  = (const float*)d_in[5];
  const float* masks = (const float*)d_in[6];
  float* out = (float*)d_out;

  char* ws = (char*)d_ws;
  const size_t wcat_b = (size_t)DOUT * KTOT * 2;   // 3,440,640
  const size_t abf_b  = (size_t)KL * DIN * 2;      //   163,840
  const size_t xcat_b = (size_t)TOK * KTOT * 2;    // 88,080,384

  unsigned short* Wcat = (unsigned short*)ws;
  unsigned short* Abf  = (unsigned short*)(ws + wcat_b);

  pack_weights<<<1024, 256, 0, stream>>>(W, A, Bw, scal, Wcat, Abf);

  if (ws_size >= wcat_b + abf_b + xcat_b) {
    unsigned short* xcat = (unsigned short*)(ws + wcat_b + abf_b);
    lora_h_fused<<<TOK / BM, 256, 0, stream>>>(x, Abf, masks, xcat);
    main_gemm_fused<<<(TOK / BM) * (DOUT / BN), 256, 0, stream>>>(xcat, Wcat, bias, out);
  } else {
    unsigned short* Hwb = (unsigned short*)(ws + wcat_b + abf_b);
    lora_h_small<<<TOK / BM, 256, 0, stream>>>(x, Abf, masks, Hwb);
    main_gemm_small<<<(TOK / BM) * (DOUT / BN), 256, 0, stream>>>(x, Wcat, Hwb, bias, out);
  }
}
